// LSTM_54992761258225
// MI455X (gfx1250) — matrix operand order repaired and checked by reading, not compiled, not run on hardware
//
#include <hip/hip_runtime.h>

// ---------------------------------------------------------------------------
// Bidirectional 2-layer LSTM + FC for MI455X (gfx1250), wave32 + WMMA.
//   B=128, T=512, I=128, H=512, G=4H=2048, D=2, O=48
// Strategy:
//   1) convert x / all weights to f16 (one-time, bandwidth-trivial)
//   2) xg = x @ W_ih^T + b_ih + b_hh  as big WMMA GEMMs (bulk of FLOPs)
//   3) persistent scan kernel: 8 batch-tiles x 2 directions; h kept in LDS,
//      per-step h @ W_hh^T via v_wmma_f32_16x16x32_f16, gate exchange via LDS,
//      cell state c in registers.
//   4) FC [128 x 524288] @ [524288 x 48] with per-wave K-split + LDS reduce.
// ---------------------------------------------------------------------------

typedef __attribute__((ext_vector_type(16))) _Float16 v16h;
typedef __attribute__((ext_vector_type(8)))  _Float16 v8h;
typedef __attribute__((ext_vector_type(8)))  float    v8f;

#define WMMA_F32_F16(a, b, c) \
  __builtin_amdgcn_wmma_f32_16x16x32_f16(false, (a), false, (b), (short)0, (c), false, false)

static constexpr int Bsz = 128;
static constexpr int Tsz = 512;
static constexpr int Isz = 128;
static constexpr int Hsz = 512;
static constexpr int Gsz = 4 * Hsz;        // 2048
static constexpr int MT  = Bsz * Tsz;      // 65536 rows for the xg GEMMs
static constexpr int Osz = 48;
static constexpr long FCK = (long)Tsz * 2 * Hsz;  // 524288

// ---------------------------------------------------------------------------
// f32 -> f16 conversion (grid-stride)
// ---------------------------------------------------------------------------
__global__ void cvt_f32_f16(const float* __restrict__ src,
                            _Float16* __restrict__ dst, long n) {
  long i = (long)blockIdx.x * blockDim.x + threadIdx.x;
  long stride = (long)gridDim.x * blockDim.x;
  for (; i < n; i += stride) dst[i] = (_Float16)src[i];
}

// ---------------------------------------------------------------------------
// Fragment loaders (CDNA5 WMMA 16-bit layouts, wave32).
// A 16x32 (MxK): lane L (L<16) row=L holds K = kb..kb+7 and 16+kb..16+kb+7,
//                kb = (L>=16)*8.  Two contiguous 16B runs per lane.
// B 32x16 (KxN) taken from row-major W[N,K]: lane holds col = L&15 and
//                K = kb..kb+7 and 16+kb..16+kb+7, kb = (L>=16)*8 (same as A).
// C/D 16x16 f32: lane L col = L&15, vgpr r -> row = r + 8*(L>=16).
// ---------------------------------------------------------------------------
__device__ __forceinline__ v16h load_a_frag(const _Float16* A, long lda,
                                            int m0, int k0, int lane) {
  int row = m0 + (lane & 15);
  int kb  = ((lane >> 4) & 1) * 8;
  const _Float16* p = A + (long)row * lda + k0 + kb;
  v8h lo = *(const v8h*)(p);
  v8h hi = *(const v8h*)(p + 16);
  v16h r;
#pragma unroll
  for (int i = 0; i < 8; ++i) { r[i] = lo[i]; r[i + 8] = hi[i]; }
  return r;
}

__device__ __forceinline__ v16h load_bT_frag(const _Float16* W, long ldw,
                                             int n0, int k0, int lane) {
  int col  = n0 + (lane & 15);
  int koff = ((lane >> 4) & 1) * 8;
  const _Float16* p = W + (long)col * ldw + k0 + koff;
  v8h lo = *(const v8h*)(p);
  v8h hi = *(const v8h*)(p + 16);
  v16h r;
#pragma unroll
  for (int i = 0; i < 8; ++i) { r[i] = lo[i]; r[i + 8] = hi[i]; }
  return r;
}

// ---------------------------------------------------------------------------
// xg = A[MT x K] @ W[G x K]^T + b_ih + b_hh   (f16 in, f32 out)
// block = 256 (8 waves); wave -> 16M x 64N; grid = (G/512, MT/16)
// ---------------------------------------------------------------------------
__global__ void __launch_bounds__(256)
wmma_xproj(const _Float16* __restrict__ A, const _Float16* __restrict__ W,
           const float* __restrict__ b1, const float* __restrict__ b2,
           float* __restrict__ out, int K) {
  const int lane = threadIdx.x & 31;
  const int wave = threadIdx.x >> 5;
  const int m0 = blockIdx.y * 16;
  const int n0 = blockIdx.x * 512 + wave * 64;

  v8f acc[4] = {};
  for (int k0 = 0; k0 < K; k0 += 32) {
    v16h a = load_a_frag(A, K, m0, k0, lane);
#pragma unroll
    for (int t = 0; t < 4; ++t) {
      v16h b = load_bT_frag(W, K, n0 + t * 16, k0, lane);
      acc[t] = WMMA_F32_F16(a, b, acc[t]);
    }
    // stream-ahead on the weight rows (global_prefetch_b8)
    __builtin_prefetch(W + (long)(n0 + (lane & 15)) * K + k0 + 64, 0, 1);
  }

  const int mbase = m0 + ((lane >> 4) & 1) * 8;
#pragma unroll
  for (int t = 0; t < 4; ++t) {
    int n = n0 + t * 16 + (lane & 15);
    float bias = b1[n] + b2[n];
#pragma unroll
    for (int r = 0; r < 8; ++r)
      out[(long)(mbase + r) * Gsz + n] = acc[t][r] + bias;
  }
}

// ---------------------------------------------------------------------------
// Persistent LSTM scan. grid = (B/16, 2 directions), block = 512 (16 waves).
// Each block: 16 batch rows, all 2048 gate columns, T sequential steps.
// h kept in LDS (f16), gates exchanged via LDS (f32), c in registers.
// wave w covers gate columns [w*128, w*128+128) -> gate id = w/4 (i,f,g,o).
// ---------------------------------------------------------------------------
__global__ void __launch_bounds__(512)
lstm_scan(const float* __restrict__ xgF, const float* __restrict__ xgB,
          const _Float16* __restrict__ whF, const _Float16* __restrict__ whB,
          _Float16* __restrict__ hcat) {
  __shared__ __align__(32) _Float16 h_sh[16 * Hsz];   // 16 KB
  __shared__ float g_sh[4 * 16 * Hsz];                // 128 KB

  const int dir = blockIdx.y;
  const float* xg = dir ? xgB : xgF;
  const _Float16* Wh = dir ? whB : whF;
  const int bm   = blockIdx.x * 16;
  const int tid  = threadIdx.x;
  const int lane = tid & 31;
  const int wave = tid >> 5;                 // 0..15
  const int gate = wave >> 2;                // 0:i 1:f 2:g 3:o
  const int ncolw = wave * 128;

  for (int i = tid; i < 16 * Hsz; i += 512) h_sh[i] = (_Float16)0.f;

  float c_reg[16];
#pragma unroll
  for (int i = 0; i < 16; ++i) c_reg[i] = 0.f;

  const int cm = tid >> 5;           // cell-update row owned by this thread
  const int jb = (tid & 31) * 16;    // cell-update col base (16 elems)
  __syncthreads();

  for (int step = 0; step < Tsz; ++step) {
    const int t = dir ? (Tsz - 1 - step) : step;

    // ---- g = h @ Wh^T (WMMA over K=512, 8 N-subtiles per wave) ----
    v8f acc[8] = {};
    for (int k0 = 0; k0 < Hsz; k0 += 32) {
      v16h a = load_a_frag(h_sh, Hsz, 0, k0, lane);   // ds_load_b128 x2
#pragma unroll
      for (int q = 0; q < 8; ++q) {
        v16h b = load_bT_frag(Wh, Hsz, ncolw + q * 16, k0, lane);
        acc[q] = WMMA_F32_F16(a, b, acc[q]);
      }
      __builtin_prefetch(Wh + (long)(ncolw + (lane & 15)) * Hsz + k0 + 64, 0, 1);
    }

    // ---- add xg, apply activation, publish to LDS ----
    const int nl = lane & 15;
    const int hi = (lane >> 4) & 1;
#pragma unroll
    for (int q = 0; q < 8; ++q) {
      const int gcol = ncolw + q * 16 + nl;  // 0..2047
      const int j    = gcol & (Hsz - 1);     // column within gate
#pragma unroll
      for (int r = 0; r < 8; ++r) {
        const int m = r + 8 * hi;
        const long brow = (long)(bm + m) * Tsz + t;
        float v = acc[q][r] + xg[brow * Gsz + gcol];
        v = (gate == 2) ? tanhf(v) : 1.f / (1.f + __expf(-v));
        g_sh[(gate * 16 + m) * Hsz + j] = v;
      }
    }
    __syncthreads();

    // ---- elementwise cell update: c = f*c + i*g ; h = o*tanh(c) ----
    {
      const long orow = ((long)(bm + cm) * Tsz + t) * (2 * Hsz) + (long)dir * Hsz;
#pragma unroll
      for (int jj = 0; jj < 16; ++jj) {
        const int j = jb + jj;
        const float ig = g_sh[(0 * 16 + cm) * Hsz + j];
        const float fg = g_sh[(1 * 16 + cm) * Hsz + j];
        const float gg = g_sh[(2 * 16 + cm) * Hsz + j];
        const float og = g_sh[(3 * 16 + cm) * Hsz + j];
        const float c = fg * c_reg[jj] + ig * gg;
        c_reg[jj] = c;
        const float h = og * tanhf(c);
        h_sh[cm * Hsz + j] = (_Float16)h;
        hcat[orow + j] = (_Float16)h;
      }
    }
    __syncthreads();
  }
}

// ---------------------------------------------------------------------------
// FC: out[128 x 48] = h1[128 x 524288] @ fcw[48 x 524288]^T + bias.
// grid = (3, 8); block = 256 (8 waves K-split) + LDS reduction.
// ---------------------------------------------------------------------------
__global__ void __launch_bounds__(256)
wmma_fc(const _Float16* __restrict__ A, const _Float16* __restrict__ W,
        const float* __restrict__ bias, float* __restrict__ out) {
  __shared__ float red[8 * 32 * 8];
  const int lane = threadIdx.x & 31;
  const int wave = threadIdx.x >> 5;
  const int n0 = blockIdx.x * 16;
  const int m0 = blockIdx.y * 16;

  v8f acc = {};
  for (long k0 = (long)wave * 32; k0 < FCK; k0 += 8 * 32) {
    v16h a = load_a_frag(A, FCK, m0, (int)k0, lane);
    v16h b = load_bT_frag(W, FCK, n0, (int)k0, lane);
    acc = WMMA_F32_F16(a, b, acc);
  }

#pragma unroll
  for (int r = 0; r < 8; ++r) red[(wave * 32 + lane) * 8 + r] = acc[r];
  __syncthreads();

  if (wave == 0) {
    const int n = n0 + (lane & 15);
#pragma unroll
    for (int r = 0; r < 8; ++r) {
      float s = 0.f;
#pragma unroll
      for (int w2 = 0; w2 < 8; ++w2) s += red[(w2 * 32 + lane) * 8 + r];
      const int m = m0 + r + 8 * ((lane >> 4) & 1);
      out[m * Osz + n] = s + bias[n];
    }
  }
}

// ---------------------------------------------------------------------------
// Host-side launch
// ---------------------------------------------------------------------------
extern "C" void kernel_launch(void* const* d_in, const int* in_sizes, int n_in,
                              void* d_out, int out_size, void* d_ws, size_t ws_size,
                              hipStream_t stream) {
  const float* x        = (const float*)d_in[0];
  const float* w_ih[4]  = {(const float*)d_in[1],  (const float*)d_in[5],
                           (const float*)d_in[9],  (const float*)d_in[13]};
  const float* w_hh[4]  = {(const float*)d_in[2],  (const float*)d_in[6],
                           (const float*)d_in[10], (const float*)d_in[14]};
  const float* b_ih[4]  = {(const float*)d_in[3],  (const float*)d_in[7],
                           (const float*)d_in[11], (const float*)d_in[15]};
  const float* b_hh[4]  = {(const float*)d_in[4],  (const float*)d_in[8],
                           (const float*)d_in[12], (const float*)d_in[16]};
  const float* fc_w = (const float*)d_in[17];
  const float* fc_b = (const float*)d_in[18];
  float* out = (float*)d_out;

  // workspace carve-out (deterministic, same every call)
  char* ws = (char*)d_ws;
  auto carve = [&](size_t bytes) -> char* {
    char* p = ws;
    ws += (bytes + 255) & ~(size_t)255;
    return p;
  };
  _Float16* x16      = (_Float16*)carve((size_t)MT * Isz * 2);
  _Float16* wih16[4]; _Float16* whh16[4];
  const long wihN[4] = {(long)Gsz * Isz, (long)Gsz * Isz,
                        (long)Gsz * 2 * Hsz, (long)Gsz * 2 * Hsz};
  for (int i = 0; i < 4; ++i) wih16[i] = (_Float16*)carve((size_t)wihN[i] * 2);
  for (int i = 0; i < 4; ++i) whh16[i] = (_Float16*)carve((size_t)Gsz * Hsz * 2);
  _Float16* fcw16   = (_Float16*)carve((size_t)Osz * FCK * 2);
  _Float16* h0cat16 = (_Float16*)carve((size_t)MT * 2 * Hsz * 2);
  _Float16* h1cat16 = (_Float16*)carve((size_t)MT * 2 * Hsz * 2);
  float*    xgf     = (float*)carve((size_t)MT * Gsz * 4);
  float*    xgb     = (float*)carve((size_t)MT * Gsz * 4);

  // 1) conversions to f16
  cvt_f32_f16<<<8192, 256, 0, stream>>>(x, x16, (long)MT * Isz);
  for (int i = 0; i < 4; ++i) {
    cvt_f32_f16<<<4096, 256, 0, stream>>>(w_ih[i], wih16[i], wihN[i]);
    cvt_f32_f16<<<4096, 256, 0, stream>>>(w_hh[i], whh16[i], (long)Gsz * Hsz);
  }
  cvt_f32_f16<<<8192, 256, 0, stream>>>(fc_w, fcw16, (long)Osz * FCK);

  // 2) layer 0 input projections (K = 128)
  {
    dim3 grid(Gsz / 512, MT / 16);
    wmma_xproj<<<grid, 256, 0, stream>>>(x16, wih16[0], b_ih[0], b_hh[0], xgf, Isz);
    wmma_xproj<<<grid, 256, 0, stream>>>(x16, wih16[1], b_ih[1], b_hh[1], xgb, Isz);
  }
  // 3) layer 0 scan (both directions concurrently via grid.y)
  lstm_scan<<<dim3(Bsz / 16, 2), 512, 0, stream>>>(xgf, xgb, whh16[0], whh16[1], h0cat16);

  // 4) layer 1 input projections (K = 1024)
  {
    dim3 grid(Gsz / 512, MT / 16);
    wmma_xproj<<<grid, 256, 0, stream>>>(h0cat16, wih16[2], b_ih[2], b_hh[2], xgf, 2 * Hsz);
    wmma_xproj<<<grid, 256, 0, stream>>>(h0cat16, wih16[3], b_ih[3], b_hh[3], xgb, 2 * Hsz);
  }
  // 5) layer 1 scan
  lstm_scan<<<dim3(Bsz / 16, 2), 512, 0, stream>>>(xgf, xgb, whh16[2], whh16[3], h1cat16);

  // 6) final FC
  wmma_fc<<<dim3(Osz / 16, Bsz / 16), 256, 0, stream>>>(h1cat16, fcw16, fc_b, out);
}
